// MinkUNetMultiResolutionDiffusion_70806830842213
// MI455X (gfx1250) — matrix operand order repaired and checked by reading, not compiled, not run on hardware
//
#include <hip/hip_runtime.h>

#define ASYNC_LDS 1

// ---------------- CDNA5 WMMA types & helpers (wave32, gfx1250) ----------------
typedef __attribute__((ext_vector_type(16))) __bf16        v16bf;
typedef __attribute__((ext_vector_type(8)))  float         v8f;
typedef __attribute__((ext_vector_type(8)))  unsigned int  v8u;

__device__ __forceinline__ v16bf mkfrag(uint4 a, uint4 b) {
  v8u u;
  u[0] = a.x; u[1] = a.y; u[2] = a.z; u[3] = a.w;
  u[4] = b.x; u[5] = b.y; u[6] = b.z; u[7] = b.w;
  return __builtin_bit_cast(v16bf, u);
}
__device__ __forceinline__ v16bf ldfrag(const unsigned* p) {
  const uint4* q = (const uint4*)p;
  return mkfrag(q[0], q[1]);
}
// D = A(16x32 bf16) * B(32x16 bf16) + C(16x16 f32)
__device__ __forceinline__ v8f wmma_bf16(v16bf a, v16bf b, v8f c) {
  return __builtin_amdgcn_wmma_f32_16x16x32_bf16(false, a, false, b, (short)0, c,
                                                 false, false);
}
__device__ __forceinline__ unsigned short f2bf(float f) {
  unsigned u = __builtin_bit_cast(unsigned, f);
  return (unsigned short)((u + 0x8000u) >> 16);
}
__device__ __forceinline__ float bf2f(unsigned short h) {
  unsigned u = ((unsigned)h) << 16;
  return __builtin_bit_cast(float, u);
}
__device__ __forceinline__ unsigned pack2(float a, float b) {
  return (unsigned)f2bf(a) | ((unsigned)f2bf(b) << 16);
}
// A-fragment from a packed-bf16 row, tile K-range starting at dword dwBase.
// A layout (contiguous): lanes<16 -> K {0..15}; lanes>=16 -> K {16..31}
__device__ __forceinline__ v16bf ld_a_dense(const unsigned* rowPtr, int dwBase, bool hi) {
  const uint4* p0 = (const uint4*)(rowPtr + dwBase + (hi ? 8 : 0));
  const uint4* p1 = (const uint4*)(rowPtr + dwBase + 4 + (hi ? 8 : 0));
  return mkfrag(p0[0], p1[0]);
}
// per-channel sum/sumsq accumulation for BN (D layout: col = lane&15 in both halves)
__device__ __forceinline__ void acc_stats(v8f acc, float* sum, float* sq, int ch, bool hi) {
  float s = 0.f, q = 0.f;
#pragma unroll
  for (int i = 0; i < 8; ++i) { float v = acc[i]; s += v; q += v * v; }
  s += __shfl_xor(s, 16);
  q += __shfl_xor(q, 16);
  if (!hi) { atomicAdd(sum + ch, s); atomicAdd(sq + ch, q); }
}
__device__ __forceinline__ void store_tile(v8f acc, unsigned short* dst, int n0, int col,
                                           int stride, bool hi) {
#pragma unroll
  for (int i = 0; i < 8; ++i) {
    int r = n0 + i + (hi ? 8 : 0);
    dst[(size_t)r * stride + col] = f2bf(acc[i]);
  }
}

// Cooperative stage of weight fragments into LDS (whole block), then barrier.
// Uses CDNA5 async global->LDS copies (ASYNCcnt) when enabled.
__device__ __forceinline__ void stage_lds(unsigned* smem, const unsigned* gsrc, int ndw) {
  int n4 = ndw >> 2;  // 16B packets
#if ASYNC_LDS
  unsigned ldsBase = (unsigned)(size_t)smem;
  for (int i = threadIdx.x; i < n4; i += blockDim.x) {
    unsigned lds = ldsBase + i * 16u;
    const unsigned* g = gsrc + i * 4;
    asm volatile("global_load_async_to_lds_b128 %0, %1, off" : : "v"(lds), "v"(g)
                 : "memory");
  }
  asm volatile("s_wait_asynccnt 0x0" : : : "memory");
#else
  uint4* s4 = (uint4*)smem;
  const uint4* g4 = (const uint4*)gsrc;
  for (int i = threadIdx.x; i < n4; i += blockDim.x) s4[i] = g4[i];
#endif
  __syncthreads();
}

// ---------------- small utility kernels ----------------
__global__ void zero_kernel(float* p, int n) {
  int i = blockIdx.x * blockDim.x + threadIdx.x;
  if (i < n) p[i] = 0.f;
}

// x [N,8] f32 -> xbf [NR rows][8] bf16 (4 dwords/row), zero pad rows >= N
__global__ void prep_x_kernel(const float* __restrict__ x, unsigned* __restrict__ xbf,
                              int N, int NR) {
  size_t total = (size_t)NR * 4;
  for (size_t i = (size_t)blockIdx.x * blockDim.x + threadIdx.x; i < total;
       i += (size_t)gridDim.x * blockDim.x) {
    size_t row = i >> 2;
    int d = (int)(i & 3);
    float a = 0.f, b = 0.f;
    if (row < (size_t)N) { a = x[row * 8 + d * 2]; b = x[row * 8 + d * 2 + 1]; }
    xbf[i] = pack2(a, b);
  }
}

// Generic weight prep: src flattened [Ktot][Cout] f32 -> B fragments.
// Tile = (kGroup, colTile); per tile 256 dwords, lane-major (8 dwords per lane):
// lane: col = ct*16 + (lane&15), kbase = g*32 + (lane>=16?16:0); dword j packs K=kbase+2j,2j+1.
__global__ void prep_w_kernel(const float* __restrict__ src, unsigned* __restrict__ dst,
                              int Ktot, int Cout) {
  int nCt = Cout >> 4;
  int tile = blockIdx.x;
  int g = tile / nCt, ct = tile % nCt;
  int lane = threadIdx.x & 31;
  int col = ct * 16 + (lane & 15);
  int kb = g * 32 + ((lane >= 16) ? 16 : 0);
  unsigned* out = dst + (size_t)tile * 256 + lane * 8;
#pragma unroll
  for (int j = 0; j < 8; ++j) {
    int k0 = kb + 2 * j, k1 = k0 + 1;
    float a = (k0 < Ktot) ? src[(size_t)k0 * Cout + col] : 0.f;
    float b = (k1 < Ktot) ? src[(size_t)k1 * Cout + col] : 0.f;
    out[j] = pack2(a, b);
  }
}

// time MLP: t = emb[time]; t = silu(t@Wt1+bt1)@Wt2+bt2 -> bf16 [B][32]
__global__ void time_mlp_kernel(const float* __restrict__ emb, const int* __restrict__ tIdx,
                                const float* __restrict__ Wt1, const float* __restrict__ bt1,
                                const float* __restrict__ Wt2, const float* __restrict__ bt2,
                                unsigned short* __restrict__ tEmb) {
  __shared__ float t0[8][32], t1[8][32];
  int b = threadIdx.x >> 5;
  int c = threadIdx.x & 31;
  t0[b][c] = emb[(size_t)tIdx[b] * 32 + c];
  __syncthreads();
  float a = bt1[c];
  for (int k = 0; k < 32; ++k) a += t0[b][k] * Wt1[k * 32 + c];
  t1[b][c] = a / (1.f + __expf(-a));  // silu
  __syncthreads();
  float o = bt2[c];
  for (int k = 0; k < 32; ++k) o += t1[b][k] * Wt2[k * 32 + c];
  tEmb[b * 32 + c] = f2bf(o);
}

// ---------------- conv5: h0raw[N,32] = sum_k xp[nb5[k]] @ W0[k] ----------------
__global__ __launch_bounds__(256) void conv5_kernel(
    const unsigned* __restrict__ xbf, const int* __restrict__ nb5,
    const unsigned* __restrict__ W0f, unsigned short* __restrict__ h0,
    float* __restrict__ s0, int N, int NR) {
  extern __shared__ unsigned smem[];           // 64 tiles * 1KB = 64KB
  stage_lds(smem, W0f, 64 * 256);
  const int lane = threadIdx.x & 31;
  const bool hi = lane >= 16;
  const int tile = blockIdx.x * (blockDim.x >> 5) + (threadIdx.x >> 5);
  const int n0 = tile * 16;
  const bool active = n0 < NR;
  const int arow = active ? n0 + (lane & 15) : 0;
  const bool rowOK = active && (arow < N);
  const uint4* xr = (const uint4*)xbf;         // 1 row of x = 8 bf16 = one uint4
  const size_t strideN = (size_t)N;
  const int* pt = nb5 + arow + (hi ? 2 * strideN : 0);  // tap pointer, +4N per group
  v8f acc0 = {};
  v8f acc1 = {};
  int tap = hi ? 2 : 0;
  for (int g = 0; g < 32; ++g) {               // 4 taps per K-group (K = 4*8 = 32)
    int i0 = N, i2 = N;                        // row N of xbf is the zero pad row
    if (rowOK) {
      if (tap < 125) i0 = pt[0];
      if (tap + 1 < 125) i2 = pt[strideN];
      if (tap + 4 < 125) __builtin_prefetch(pt + 4 * strideN, 0, 1);
    }
    pt += 4 * strideN;
    tap += 4;
    v16bf A = mkfrag(xr[i0], xr[i2]);
    v16bf B0 = ldfrag(smem + (g * 2 + 0) * 256 + lane * 8);
    v16bf B1 = ldfrag(smem + (g * 2 + 1) * 256 + lane * 8);
    acc0 = wmma_bf16(A, B0, acc0);
    acc1 = wmma_bf16(A, B1, acc1);
  }
  if (!active) return;
  const int ch = lane & 15;
  acc_stats(acc0, s0, s0 + 32, ch, hi);
  acc_stats(acc1, s0, s0 + 32, ch + 16, hi);
  store_tile(acc0, h0, n0, ch, 32, hi);
  store_tile(acc1, h0, n0, ch + 16, 32, hi);
}

// BN(h0)+ReLU, concat time embedding -> hcat [NR,64] bf16 (zeros beyond N)
__global__ void norm_cat_kernel(const unsigned short* __restrict__ h0,
                                const unsigned short* __restrict__ tEmb,
                                const int* __restrict__ bidx, const float* __restrict__ s0,
                                const float* __restrict__ g0, const float* __restrict__ b0,
                                unsigned short* __restrict__ hcat, int Nv, int NR, float invN) {
  __shared__ float sc[32], sh[32];
  if (threadIdx.x < 32) {
    int c = threadIdx.x;
    float m = s0[c] * invN;
    float var = s0[32 + c] * invN - m * m;
    float s = g0[c] * rsqrtf(var + 1e-5f);
    sc[c] = s;
    sh[c] = b0[c] - m * s;
  }
  __syncthreads();
  size_t total = (size_t)NR * 64;
  for (size_t i = (size_t)blockIdx.x * blockDim.x + threadIdx.x; i < total;
       i += (size_t)gridDim.x * blockDim.x) {
    int c = (int)(i & 63);
    size_t row = i >> 6;
    float v = 0.f;
    if (row < (size_t)Nv) {
      if (c < 32) {
        v = bf2f(h0[row * 32 + c]) * sc[c] + sh[c];
        v = v > 0.f ? v : 0.f;
      } else {
        v = bf2f(tEmb[(size_t)bidx[row] * 32 + (c - 32)]);
      }
    }
    hcat[i] = f2bf(v);
  }
}

// fused: r1raw = hcat @ Wb1 (K=64,Cout=16)  and  hdsraw = hcat @ Wds (K=64,Cout=64)
__global__ __launch_bounds__(256) void bneck_a_kernel(
    const unsigned* __restrict__ hcat, const unsigned* __restrict__ Wb1f,
    const unsigned* __restrict__ Wdsf, unsigned short* __restrict__ r1,
    unsigned short* __restrict__ hds, float* __restrict__ s1, float* __restrict__ sds,
    int NR) {
  extern __shared__ unsigned smem[];           // 512 + 2048 dwords
  unsigned* sWb1 = smem;
  unsigned* sWds = smem + 512;
  stage_lds(sWb1, Wb1f, 512);
  stage_lds(sWds, Wdsf, 2048);
  const int lane = threadIdx.x & 31;
  const bool hi = lane >= 16;
  const int tile = blockIdx.x * (blockDim.x >> 5) + (threadIdx.x >> 5);
  const int n0 = tile * 16;
  const bool active = n0 < NR;
  const int arow = active ? n0 + (lane & 15) : 0;
  const unsigned* rowPtr = hcat + (size_t)arow * 32;  // 64 bf16 = 32 dwords/row
  v8f ar = {};
  v8f ads[4] = {};
#pragma unroll
  for (int kt = 0; kt < 2; ++kt) {
    v16bf A = ld_a_dense(rowPtr, kt * 16, hi);
    ar = wmma_bf16(A, ldfrag(sWb1 + kt * 256 + lane * 8), ar);
#pragma unroll
    for (int ct = 0; ct < 4; ++ct)
      ads[ct] = wmma_bf16(A, ldfrag(sWds + (kt * 4 + ct) * 256 + lane * 8), ads[ct]);
  }
  if (!active) return;
  const int ch = lane & 15;
  acc_stats(ar, s1, s1 + 16, ch, hi);
  store_tile(ar, r1, n0, ch, 16, hi);
#pragma unroll
  for (int ct = 0; ct < 4; ++ct) {
    acc_stats(ads[ct], sds, sds + 64, ct * 16 + ch, hi);
    store_tile(ads[ct], hds, n0, ct * 16 + ch, 64, hi);
  }
}

// generic BN+ReLU (bf16 -> bf16), zero pad rows >= Nv up to NR
__global__ void bn_relu_kernel(const unsigned short* __restrict__ src,
                               unsigned short* __restrict__ dst,
                               const float* __restrict__ stats, const float* __restrict__ g,
                               const float* __restrict__ b, int C, int Nv, int NR,
                               float invN) {
  __shared__ float sc[64], sh[64];
  if ((int)threadIdx.x < C) {
    int c = threadIdx.x;
    float m = stats[c] * invN;
    float var = stats[C + c] * invN - m * m;
    float s = g[c] * rsqrtf(var + 1e-5f);
    sc[c] = s;
    sh[c] = b[c] - m * s;
  }
  __syncthreads();
  size_t total = (size_t)NR * C;
  for (size_t i = (size_t)blockIdx.x * blockDim.x + threadIdx.x; i < total;
       i += (size_t)gridDim.x * blockDim.x) {
    int c = (int)(i % (size_t)C);
    size_t row = i / (size_t)C;
    float v = 0.f;
    if (row < (size_t)Nv) {
      v = bf2f(src[i]) * sc[c] + sh[c];
      v = v > 0.f ? v : 0.f;
    }
    dst[i] = f2bf(v);
  }
}

// conv3: r2raw = sum_k r1n[nb3[k]] @ Wb2[k]  (27 taps * 16ch => 14 K-groups)
__global__ __launch_bounds__(256) void conv3_kernel(
    const unsigned* __restrict__ r1n, const int* __restrict__ nb3,
    const unsigned* __restrict__ Wb2f, unsigned short* __restrict__ r2,
    float* __restrict__ s2, int N, int NR) {
  extern __shared__ unsigned smem[];           // 14 * 256 dwords
  stage_lds(smem, Wb2f, 14 * 256);
  const int lane = threadIdx.x & 31;
  const bool hi = lane >= 16;
  const int tile = blockIdx.x * (blockDim.x >> 5) + (threadIdx.x >> 5);
  const int n0 = tile * 16;
  const bool active = n0 < NR;
  const int arow = active ? n0 + (lane & 15) : 0;
  const bool rowOK = active && (arow < N);
  const size_t strideN = (size_t)N;
  const int* pt = nb3 + arow;
  v8f acc = {};
  for (int g = 0; g < 14; ++g) {               // 2 taps per K-group
    int i0 = N, i1 = N;
    if (rowOK) {
      i0 = pt[0];
      if (2 * g + 1 < 27) i1 = pt[strideN];
    }
    pt += 2 * strideN;
    const uint4* p0 = (const uint4*)(r1n + (size_t)(hi ? i1 : i0) * 8);
    const uint4* p1 = (const uint4*)(r1n + (size_t)(hi ? i1 : i0) * 8 + 4);
    v16bf A = mkfrag(p0[0], p1[0]);
    acc = wmma_bf16(A, ldfrag(smem + g * 256 + lane * 8), acc);
  }
  if (!active) return;
  const int ch = lane & 15;
  acc_stats(acc, s2, s2 + 16, ch, hi);
  store_tile(acc, r2, n0, ch, 16, hi);
}

// 1x1 expand: r3raw = r2n @ Wb3 (K=16 padded to 32, Cout=64)
__global__ __launch_bounds__(256) void bneck_c_kernel(
    const unsigned* __restrict__ r2n, const unsigned* __restrict__ Wb3f,
    unsigned short* __restrict__ r3, float* __restrict__ s3, int NR) {
  extern __shared__ unsigned smem[];           // 4 * 256 dwords
  stage_lds(smem, Wb3f, 4 * 256);
  const int lane = threadIdx.x & 31;
  const bool hi = lane >= 16;
  const int tile = blockIdx.x * (blockDim.x >> 5) + (threadIdx.x >> 5);
  const int n0 = tile * 16;
  const bool active = n0 < NR;
  const int arow = active ? n0 + (lane & 15) : 0;
  const uint4* p = (const uint4*)(r2n + (size_t)arow * 8);
  uint4 z; z.x = z.y = z.z = z.w = 0u;
  v16bf A = mkfrag(hi ? z : p[0], hi ? z : p[1]);  // K 16..31 (hi lanes) are zero padding
  v8f acc[4] = {};
#pragma unroll
  for (int ct = 0; ct < 4; ++ct)
    acc[ct] = wmma_bf16(A, ldfrag(smem + ct * 256 + lane * 8), acc[ct]);
  if (!active) return;
  const int ch = lane & 15;
#pragma unroll
  for (int ct = 0; ct < 4; ++ct) {
    acc_stats(acc[ct], s3, s3 + 64, ct * 16 + ch, hi);
    store_tile(acc[ct], r3, n0, ct * 16 + ch, 64, hi);
  }
}

// h2 = relu(bn3(r3raw) + bnds(hdsraw)), zeros beyond N (pad row for conv2 gather)
__global__ void residual_kernel(const unsigned short* __restrict__ r3,
                                const unsigned short* __restrict__ hds,
                                const float* __restrict__ s3, const float* __restrict__ sds,
                                const float* __restrict__ g3, const float* __restrict__ b3,
                                const float* __restrict__ gds, const float* __restrict__ bds,
                                unsigned short* __restrict__ h2, int Nv, int NR, float invN) {
  __shared__ float sc3[64], sh3[64], scd[64], shd[64];
  if (threadIdx.x < 64) {
    int c = threadIdx.x;
    float m = s3[c] * invN, var = s3[64 + c] * invN - m * m;
    float s = g3[c] * rsqrtf(var + 1e-5f);
    sc3[c] = s; sh3[c] = b3[c] - m * s;
    m = sds[c] * invN; var = sds[64 + c] * invN - m * m;
    s = gds[c] * rsqrtf(var + 1e-5f);
    scd[c] = s; shd[c] = bds[c] - m * s;
  }
  __syncthreads();
  size_t total = (size_t)NR * 64;
  for (size_t i = (size_t)blockIdx.x * blockDim.x + threadIdx.x; i < total;
       i += (size_t)gridDim.x * blockDim.x) {
    int c = (int)(i & 63);
    size_t row = i >> 6;
    float v = 0.f;
    if (row < (size_t)Nv) {
      v = bf2f(r3[i]) * sc3[c] + sh3[c] + bf2f(hds[i]) * scd[c] + shd[c];
      v = v > 0.f ? v : 0.f;
    }
    h2[i] = f2bf(v);
  }
}

// conv2 (k=2,s=2): outraw[M,64] = sum_k h2[nb2[k]] @ Wd[k]  (8 taps * 64ch => 16 groups)
__global__ __launch_bounds__(256) void conv2_kernel(
    const unsigned* __restrict__ h2, const int* __restrict__ nb2,
    const unsigned* __restrict__ Wdf, unsigned short* __restrict__ outraw,
    float* __restrict__ sd, int M, int MR, int padRow) {
  extern __shared__ unsigned smem[];           // 64 tiles * 1KB = 64KB
  stage_lds(smem, Wdf, 64 * 256);
  const int lane = threadIdx.x & 31;
  const bool hi = lane >= 16;
  const int tile = blockIdx.x * (blockDim.x >> 5) + (threadIdx.x >> 5);
  const int n0 = tile * 16;
  const bool active = n0 < MR;
  const int arow = active ? n0 + (lane & 15) : 0;
  const bool rowOK = active && (arow < M);
  v8f acc[4] = {};
  for (int g = 0; g < 16; ++g) {               // one group = half a tap's 64 channels
    int tap = g >> 1, half = g & 1;
    int idx = padRow;
    if (rowOK) idx = nb2[(size_t)tap * M + arow];
    v16bf A = ld_a_dense(h2 + (size_t)idx * 32, half * 16, hi);
#pragma unroll
    for (int ct = 0; ct < 4; ++ct)
      acc[ct] = wmma_bf16(A, ldfrag(smem + (g * 4 + ct) * 256 + lane * 8), acc[ct]);
  }
  if (!active) return;
  const int ch = lane & 15;
#pragma unroll
  for (int ct = 0; ct < 4; ++ct) {
    acc_stats(acc[ct], sd, sd + 64, ct * 16 + ch, hi);
    store_tile(acc[ct], outraw, n0, ct * 16 + ch, 64, hi);
  }
}

// final: out f32 = relu(bn(outraw))
__global__ void bn_relu_out_kernel(const unsigned short* __restrict__ src,
                                   float* __restrict__ out, const float* __restrict__ stats,
                                   const float* __restrict__ g, const float* __restrict__ b,
                                   int M, float invM) {
  __shared__ float sc[64], sh[64];
  if (threadIdx.x < 64) {
    int c = threadIdx.x;
    float m = stats[c] * invM;
    float var = stats[64 + c] * invM - m * m;
    float s = g[c] * rsqrtf(var + 1e-5f);
    sc[c] = s;
    sh[c] = b[c] - m * s;
  }
  __syncthreads();
  size_t total = (size_t)M * 64;
  for (size_t i = (size_t)blockIdx.x * blockDim.x + threadIdx.x; i < total;
       i += (size_t)gridDim.x * blockDim.x) {
    int c = (int)(i & 63);
    float v = bf2f(src[i]) * sc[c] + sh[c];
    out[i] = v > 0.f ? v : 0.f;
  }
}

// ---------------- host side ----------------
extern "C" void kernel_launch(void* const* d_in, const int* in_sizes, int n_in,
                              void* d_out, int out_size, void* d_ws, size_t ws_size,
                              hipStream_t stream) {
  (void)n_in; (void)out_size; (void)ws_size;
  const float* x    = (const float*)d_in[0];
  const float* W0   = (const float*)d_in[1];
  const float* g0   = (const float*)d_in[2];
  const float* b0   = (const float*)d_in[3];
  const float* emb  = (const float*)d_in[4];
  const float* Wt1  = (const float*)d_in[5];
  const float* bt1  = (const float*)d_in[6];
  const float* Wt2  = (const float*)d_in[7];
  const float* bt2  = (const float*)d_in[8];
  const float* Wb1  = (const float*)d_in[9];
  const float* gb1  = (const float*)d_in[10];
  const float* bb1  = (const float*)d_in[11];
  const float* Wb2  = (const float*)d_in[12];
  const float* gb2  = (const float*)d_in[13];
  const float* bb2  = (const float*)d_in[14];
  const float* Wb3  = (const float*)d_in[15];
  const float* gb3  = (const float*)d_in[16];
  const float* bb3  = (const float*)d_in[17];
  const float* Wds  = (const float*)d_in[18];
  const float* gds  = (const float*)d_in[19];
  const float* bds  = (const float*)d_in[20];
  const float* Wd   = (const float*)d_in[21];
  const float* gd   = (const float*)d_in[22];
  const float* bd   = (const float*)d_in[23];
  const int* tIdx   = (const int*)d_in[24];
  const int* bidx   = (const int*)d_in[25];
  const int* nb5    = (const int*)d_in[26];
  const int* nb3    = (const int*)d_in[27];
  const int* nb2    = (const int*)d_in[28];

  const int N  = in_sizes[0] / 8;
  const int B  = in_sizes[24] > 8 ? 8 : in_sizes[24];
  const int M  = in_sizes[28] / 8;
  const int NR = ((N + 16 + 15) / 16) * 16;  // >= N+1 pad rows, multiple of 16
  const int MR = ((M + 15) / 16) * 16;

  char* ws = (char*)d_ws;
  size_t off = 0;
  auto alloc = [&](size_t bytes) -> void* {
    void* p = ws + off;
    off += (bytes + 255) & ~(size_t)255;
    return p;
  };
  float* stats = (float*)alloc(512 * sizeof(float));
  float* s0  = stats;        // 32 sum + 32 sumsq
  float* s1  = stats + 64;   // 16 + 16
  float* s2  = stats + 96;   // 16 + 16
  float* s3  = stats + 128;  // 64 + 64
  float* sds = stats + 256;  // 64 + 64
  float* sd  = stats + 384;  // 64 + 64
  unsigned* W0f  = (unsigned*)alloc(64 * 1024);
  unsigned* Wb1f = (unsigned*)alloc(2 * 1024);
  unsigned* Wdsf = (unsigned*)alloc(8 * 1024);
  unsigned* Wb2f = (unsigned*)alloc(14 * 1024);
  unsigned* Wb3f = (unsigned*)alloc(4 * 1024);
  unsigned* Wdf  = (unsigned*)alloc(64 * 1024);
  unsigned short* tEmb = (unsigned short*)alloc(8 * 32 * 2);
  unsigned* xbf        = (unsigned*)alloc((size_t)NR * 16);
  unsigned short* h0   = (unsigned short*)alloc((size_t)NR * 64);
  unsigned short* hcat = (unsigned short*)alloc((size_t)NR * 128);
  unsigned short* r1   = (unsigned short*)alloc((size_t)NR * 32);
  unsigned short* r1n  = (unsigned short*)alloc((size_t)NR * 32);
  unsigned short* r2   = (unsigned short*)alloc((size_t)NR * 32);
  unsigned short* r2n  = (unsigned short*)alloc((size_t)NR * 32);
  unsigned short* r3   = (unsigned short*)alloc((size_t)NR * 128);
  unsigned short* hds  = (unsigned short*)alloc((size_t)NR * 128);
  unsigned short* h2   = (unsigned short*)alloc((size_t)NR * 128);
  unsigned short* oraw = (unsigned short*)alloc((size_t)MR * 128);

  const float invN = 1.f / (float)N;
  const float invM = 1.f / (float)M;
  const int tilesN = NR / 16;
  const int tilesM = MR / 16;

  zero_kernel<<<1, 512, 0, stream>>>(stats, 512);
  prep_x_kernel<<<512, 256, 0, stream>>>(x, xbf, N, NR);
  prep_w_kernel<<<64, 32, 0, stream>>>(W0, W0f, 1000, 32);   // 125*8  K, 32 cout
  prep_w_kernel<<<2, 32, 0, stream>>>(Wb1, Wb1f, 64, 16);
  prep_w_kernel<<<8, 32, 0, stream>>>(Wds, Wdsf, 64, 64);
  prep_w_kernel<<<14, 32, 0, stream>>>(Wb2, Wb2f, 432, 16);  // 27*16 K
  prep_w_kernel<<<4, 32, 0, stream>>>(Wb3, Wb3f, 16, 64);
  prep_w_kernel<<<64, 32, 0, stream>>>(Wd, Wdf, 512, 64);    // 8*64 K
  time_mlp_kernel<<<1, 32 * B, 0, stream>>>(emb, tIdx, Wt1, bt1, Wt2, bt2, tEmb);

  conv5_kernel<<<(tilesN + 7) / 8, 256, 65536, stream>>>(xbf, nb5, W0f, h0, s0, N, NR);
  norm_cat_kernel<<<2048, 256, 0, stream>>>(h0, tEmb, bidx, s0, g0, b0, hcat, N, NR, invN);
  bneck_a_kernel<<<(tilesN + 7) / 8, 256, 10240, stream>>>((const unsigned*)hcat, Wb1f, Wdsf,
                                                           r1, hds, s1, sds, NR);
  bn_relu_kernel<<<1024, 256, 0, stream>>>(r1, r1n, s1, gb1, bb1, 16, N, NR, invN);
  conv3_kernel<<<(tilesN + 7) / 8, 256, 14336, stream>>>((const unsigned*)r1n, nb3, Wb2f,
                                                         r2, s2, N, NR);
  bn_relu_kernel<<<1024, 256, 0, stream>>>(r2, r2n, s2, gb2, bb2, 16, N, NR, invN);
  bneck_c_kernel<<<(tilesN + 7) / 8, 256, 4096, stream>>>((const unsigned*)r2n, Wb3f, r3, s3, NR);
  residual_kernel<<<2048, 256, 0, stream>>>(r3, hds, s3, sds, gb3, bb3, gds, bds, h2, N, NR, invN);
  conv2_kernel<<<(tilesM + 7) / 8, 256, 65536, stream>>>((const unsigned*)h2, nb2, Wdf,
                                                         oraw, sd, M, MR, N);
  bn_relu_out_kernel<<<1024, 256, 0, stream>>>(oraw, (float*)d_out, sd, gd, bd, M, invM);
}
